// VisualSemanticEncoder_47347719471324
// MI455X (gfx1250) — matrix operand order repaired and checked by reading, not compiled, not run on hardware
//
#include <hip/hip_runtime.h>
#include <hip/hip_bf16.h>

// ---------------------------------------------------------------------------
// VisualSemanticEncoder on MI455X (gfx1250, wave32, WMMA).
//
// Pipeline (all GEMMs via v_wmma_f32_16x16x32_bf16, f32 accumulation):
//   xb   = bf16(concat(vis, sem))                [32,768,512]
//   h1   = relu(xb @ e1_w + e1_b)                [32,768,128]   (Bt = e1_w^T)
//   h2   = relu(h1 @ e2_w + e2_b)                [32,768,128]   (Bt = e2_w^T)
//   adj  = h2 @ h2^T                             [32,768,768]   (Bt = h2!)
//   supT = (xb @ gc_w)^T                         [32,512,768]   (Bt = gc_w^T)
//   out  = mean_n relu(adj @ sup + gc_b)         [32,512]  (fused reduce epilogue)
//
// All dims/strides are template constants so address math folds into
// immediate offsets; each wave owns a 32x64 tile (2x4 fragments, 8 WMMA
// per K-step of 32).
// ---------------------------------------------------------------------------

typedef __attribute__((ext_vector_type(16))) __bf16 v16bf;
typedef __attribute__((ext_vector_type(8)))  __bf16 v8bf;
typedef __attribute__((ext_vector_type(8)))  float  v8f;

#define BS    32
#define NIMG  512
#define NKNOW 256
#define NN    768            // NIMG + NKNOW
#define DD    512
#define DH    128

// ---------------------------------------------------------------------------
// Small prep kernels
// ---------------------------------------------------------------------------
__global__ void vse_zero_f32(float* __restrict__ p, int n) {
  int i = blockIdx.x * blockDim.x + threadIdx.x;
  if (i < n) p[i] = 0.0f;
}

// concat(vis, sem) along node dim, convert f32 -> bf16
__global__ void vse_cvt_concat_x(const float* __restrict__ vis,
                                 const float* __restrict__ sem,
                                 __bf16* __restrict__ xb) {
  const int total = BS * NN * DD;
  int idx = blockIdx.x * blockDim.x + threadIdx.x;
  if (idx >= total) return;
  int d = idx & (DD - 1);
  int n = (idx >> 9) % NN;          // DD == 512 == 1<<9
  int b = idx / (NN * DD);
  float v = (n < NIMG) ? vis[((size_t)b * NIMG + n) * DD + d]
                       : sem[((size_t)b * NKNOW + (n - NIMG)) * DD + d];
  xb[idx] = (__bf16)v;
}

// in f32 [R,C] row-major -> out bf16 [C,R] row-major (i.e. transposed)
__global__ void vse_cvt_transpose(const float* __restrict__ in,
                                  __bf16* __restrict__ out, int R, int C) {
  int idx = blockIdx.x * blockDim.x + threadIdx.x;
  if (idx >= R * C) return;
  int c = idx % C;
  int r = idx / C;
  out[(size_t)c * R + r] = (__bf16)in[idx];
}

// ---------------------------------------------------------------------------
// Batched WMMA GEMM: C = act(A @ Bt^T + bias), all shapes compile-time.
//   A  : bf16 [BS?, M, K] row-major   (batch stride SA elements; 0 = shared)
//   Bt : bf16 [BS?, N, K] row-major   (batch stride SB; 0 = shared weight)
// OUT_MODE 0: store bf16 C [M,N]       (batch stride SC)
// OUT_MODE 1: store bf16 C^T [N,M]     (batch stride SC) -- packed b128 stores
// OUT_MODE 2: column-mean reduce: atomicAdd(outRed[b,n], (1/M) * sum_m act)
//
// One wave computes a 32x64 macro-tile: 2x4 fragments of 16x16, 8 WMMAs per
// K-step of 32. Tile selection is wave-uniform so EXEC is all-ones at every
// WMMA (ISA requirement).
// ---------------------------------------------------------------------------
template <int M, int N, int K, long SA, long SB, long SC,
          int OUT_MODE, bool RELU, bool BIAS>
__global__ __launch_bounds__(256)
void vse_gemm_bt(const __bf16* __restrict__ A, const __bf16* __restrict__ Bt,
                 const float* __restrict__ bias, __bf16* __restrict__ C,
                 float* __restrict__ outRed) {
  constexpr int TI = 2;                 // 32 rows per wave tile
  constexpr int TJ = 4;                 // 64 cols per wave tile
  constexpr int tilesM = M / (16 * TI);
  constexpr int tilesN = N / (16 * TJ);
  constexpr int total  = BS * tilesM * tilesN;

  const int wave = threadIdx.x >> 5;
  const int lane = threadIdx.x & 31;

  int t = blockIdx.x * 8 + wave;        // 8 waves per 256-thread block
  if (t >= total) return;               // wave-uniform exit

  const int b   = t / (tilesM * tilesN);
  const int rem = t % (tilesM * tilesN);
  const int m0  = (rem / tilesN) * (16 * TI);
  const int n0  = (rem % tilesN) * (16 * TJ);

  const int lrow = lane & 15;           // row/col within 16-wide group
  const int lhi  = lane >> 4;           // 0: lanes 0-15, 1: lanes 16-31

  const __bf16* Ab  = A  + (size_t)b * (size_t)SA;
  const __bf16* Btb = Bt + (size_t)b * (size_t)SB;

  const v8f vzero = {};
  v8f acc[TI][TJ];
#pragma unroll
  for (int i = 0; i < TI; ++i)
#pragma unroll
    for (int j = 0; j < TJ; ++j) acc[i][j] = vzero;

  // Per-lane base pointers; K-step advances by 32 elements (64 bytes).
  const __bf16* pa = Ab  + (size_t)(m0 + lrow) * K + 8 * lhi;
  const __bf16* pb = Btb + (size_t)(n0 + lrow) * K + 8 * lhi;

#pragma unroll 2
  for (int k0 = 0; k0 < K; k0 += 32) {
    v16bf af[TI], bf[TJ];

    // A fragment (16x32, MxK): lane grp 0 holds K 0-7 & 16-23, grp 1 holds
    // K 8-15 & 24-31 -> two contiguous 16B loads per lane.
#pragma unroll
    for (int i = 0; i < TI; ++i) {
      const __bf16* p = pa + (size_t)(16 * i) * K + k0;
      v8bf lo = *(const v8bf*)p;
      v8bf hi = *(const v8bf*)(p + 16);
      __builtin_prefetch((const void*)(p + 32), 0, 2);   // next K-chunk
#pragma unroll
      for (int e = 0; e < 8; ++e) { af[i][e] = lo[e]; af[i][e + 8] = hi[e]; }
    }

    // B fragment (32x16, KxN) from Bt [N,K]: lane grp 0 holds K 0-7 & 16-23,
    // grp 1 holds K 8-15 & 24-31 -> two contiguous 16B loads per lane.
#pragma unroll
    for (int j = 0; j < TJ; ++j) {
      const __bf16* p = pb + (size_t)(16 * j) * K + k0;
      v8bf lo = *(const v8bf*)p;
      v8bf hi = *(const v8bf*)(p + 16);
#pragma unroll
      for (int e = 0; e < 8; ++e) { bf[j][e] = lo[e]; bf[j][e + 8] = hi[e]; }
    }

#pragma unroll
    for (int i = 0; i < TI; ++i)
#pragma unroll
      for (int j = 0; j < TJ; ++j)
        acc[i][j] = __builtin_amdgcn_wmma_f32_16x16x32_bf16(
            false, af[i], false, bf[j], (short)0, acc[i][j], false, false);
  }

  // Epilogue. C/D fragment layout: VGPR r, lane l -> M = r + 8*(l>>4),
  // N = l&15 (within a 16x16 tile).
  if (OUT_MODE == 2) {
    constexpr float redScale = 1.0f / (float)M;
#pragma unroll
    for (int j = 0; j < TJ; ++j) {
      const int n  = n0 + 16 * j + lrow;
      const float bv = BIAS ? bias[n] : 0.0f;
      float s = 0.0f;
#pragma unroll
      for (int i = 0; i < TI; ++i)
#pragma unroll
        for (int r = 0; r < 8; ++r) {
          float v = acc[i][j][r] + bv;
          if (RELU) v = v > 0.0f ? v : 0.0f;
          s += v;
        }
      atomicAdd(outRed + (size_t)b * N + n, s * redScale);
    }
  } else if (OUT_MODE == 1) {
    // Transposed store: per lane the 8 accumulator rows are consecutive in
    // C^T's fast dim -> pack into one 16B store per fragment.
#pragma unroll
    for (int i = 0; i < TI; ++i)
#pragma unroll
      for (int j = 0; j < TJ; ++j) {
        const int n = n0 + 16 * j + lrow;
        const int m = m0 + 16 * i + 8 * lhi;
        v8bf pack;
#pragma unroll
        for (int r = 0; r < 8; ++r) {
          float v = acc[i][j][r];
          if (RELU) v = v > 0.0f ? v : 0.0f;
          pack[r] = (__bf16)v;
        }
        *(v8bf*)(C + (size_t)b * (size_t)SC + (size_t)n * M + m) = pack;
      }
  } else {
#pragma unroll
    for (int i = 0; i < TI; ++i)
#pragma unroll
      for (int j = 0; j < TJ; ++j) {
        const int n  = n0 + 16 * j + lrow;
        const float bv = BIAS ? bias[n] : 0.0f;
        __bf16* cp = C + (size_t)b * (size_t)SC +
                     (size_t)(m0 + 16 * i + 8 * lhi) * N + n;
#pragma unroll
        for (int r = 0; r < 8; ++r) {   // row stride N is a constant -> imm offsets
          float v = acc[i][j][r] + bv;
          if (RELU) v = v > 0.0f ? v : 0.0f;
          cp[(size_t)r * N] = (__bf16)v;
        }
      }
  }
}

// ---------------------------------------------------------------------------
// Host launcher
// ---------------------------------------------------------------------------
template <int M, int N>
static inline int gemmBlocks() {
  int waves = BS * (M / 32) * (N / 64);
  return (waves + 7) / 8;              // 8 waves (256 threads) per block
}

extern "C" void kernel_launch(void* const* d_in, const int* in_sizes, int n_in,
                              void* d_out, int out_size, void* d_ws, size_t ws_size,
                              hipStream_t stream) {
  (void)in_sizes; (void)n_in; (void)out_size; (void)ws_size;

  const float* vis  = (const float*)d_in[0];
  const float* sem  = (const float*)d_in[1];
  const float* e1_w = (const float*)d_in[2];
  const float* e1_b = (const float*)d_in[3];
  const float* e2_w = (const float*)d_in[4];
  const float* e2_b = (const float*)d_in[5];
  const float* gc_w = (const float*)d_in[6];
  const float* gc_b = (const float*)d_in[7];
  float* out = (float*)d_out;

  // Workspace layout (bf16 elements; ~97 MiB total, L2-resident on MI455X)
  char* ws = (char*)d_ws;
  size_t off = 0;
  __bf16* xb   = (__bf16*)(ws + off); off += (size_t)BS * NN * DD * 2;   // 25.2 MB
  __bf16* e1wT = (__bf16*)(ws + off); off += (size_t)DH * DD * 2;        // [DH,DD]
  __bf16* e2wT = (__bf16*)(ws + off); off += (size_t)DH * DH * 2;
  __bf16* gcwT = (__bf16*)(ws + off); off += (size_t)DD * DD * 2;        // [DD,DD]
  __bf16* h1b  = (__bf16*)(ws + off); off += (size_t)BS * NN * DH * 2;
  __bf16* h2b  = (__bf16*)(ws + off); off += (size_t)BS * NN * DH * 2;
  __bf16* adjb = (__bf16*)(ws + off); off += (size_t)BS * NN * NN * 2;   // 37.7 MB
  __bf16* supT = (__bf16*)(ws + off); off += (size_t)BS * DD * NN * 2;   // 25.2 MB

  // --- prep: zero output (it is poisoned), convert/concat/transpose ---
  {
    int n = BS * DD;
    vse_zero_f32<<<(n + 255) / 256, 256, 0, stream>>>(out, n);
  }
  {
    int n = BS * NN * DD;
    vse_cvt_concat_x<<<(n + 255) / 256, 256, 0, stream>>>(vis, sem, xb);
  }
  vse_cvt_transpose<<<(DD * DH + 255) / 256, 256, 0, stream>>>(e1_w, e1wT, DD, DH);
  vse_cvt_transpose<<<(DH * DH + 255) / 256, 256, 0, stream>>>(e2_w, e2wT, DH, DH);
  vse_cvt_transpose<<<(DD * DD + 255) / 256, 256, 0, stream>>>(gc_w, gcwT, DD, DD);

  // --- h1 = relu(xb @ e1_w + e1_b) : M=768, N=128, K=512 ---
  vse_gemm_bt<NN, DH, DD, (long)NN * DD, 0L, (long)NN * DH, 0, true, true>
      <<<gemmBlocks<NN, DH>(), 256, 0, stream>>>(xb, e1wT, e1_b, h1b, nullptr);

  // --- h2 = relu(h1 @ e2_w + e2_b) : M=768, N=128, K=128 ---
  vse_gemm_bt<NN, DH, DH, (long)NN * DH, 0L, (long)NN * DH, 0, true, true>
      <<<gemmBlocks<NN, DH>(), 256, 0, stream>>>(h1b, e2wT, e2_b, h2b, nullptr);

  // --- adj = h2 @ h2^T : M=768, N=768, K=128 (Bt == h2, no transpose) ---
  vse_gemm_bt<NN, NN, DH, (long)NN * DH, (long)NN * DH, (long)NN * NN, 0, false, false>
      <<<gemmBlocks<NN, NN>(), 256, 0, stream>>>(h2b, h2b, nullptr, adjb, nullptr);

  // --- supT = (xb @ gc_w)^T : M=768, N=512, K=512, packed transposed store ---
  vse_gemm_bt<NN, DD, DD, (long)NN * DD, 0L, (long)DD * NN, 1, false, false>
      <<<gemmBlocks<NN, DD>(), 256, 0, stream>>>(xb, gcwT, nullptr, supT, nullptr);

  // --- out = mean_n relu(adj @ sup + gc_b) : M=768, N=512, K=768, fused reduce ---
  vse_gemm_bt<NN, DD, NN, (long)NN * NN, (long)DD * NN, 0L, 2, true, true>
      <<<gemmBlocks<NN, DD>(), 256, 0, stream>>>(adjb, supT, gc_b, nullptr, out);
}
